// LMM_23613730194029
// MI455X (gfx1250) — compile-verified, matrix operand order checked by reading, not run on hardware
//
#include <hip/hip_runtime.h>
#include <hip/hip_bf16.h>

// ---------------------------------------------------------------------------
// Problem constants (from reference setup_inputs)
// ---------------------------------------------------------------------------
constexpr int kB = 4;
constexpr int kL = 1024;
constexpr int kN = 16384;
constexpr int kD = 256;
constexpr int kTopK = 5;
constexpr int kNChunks = 8;                 // N split for parallelism
constexpr int kChunk = kN / kNChunks;       // 2048 columns per block
constexpr int kNPart = kNChunks * 2;        // 2 half-row partial lists per chunk
constexpr int kTiles = kChunk / 16;         // 128 16-column tiles per chunk
constexpr int kWavesPerBlk = 4;             // 4 L-tiles share each staged B tile

// LDS staging: one B tile = 16 rows x 256 f16 = 512B/row; TDM pads +16B/row.
constexpr int kRowDW   = 132;               // 128 data DW + 4 pad DW
constexpr int kStageDW = 16 * kRowDW;       // 2112 DW = 8448 B per buffer

typedef __attribute__((ext_vector_type(16))) _Float16 v16h;
typedef __attribute__((ext_vector_type(8)))  _Float16 v8h;
typedef __attribute__((ext_vector_type(8)))  float    v8f;
typedef __attribute__((ext_vector_type(2)))  float    v2f;
typedef __attribute__((ext_vector_type(4)))  unsigned int v4u;
typedef __attribute__((ext_vector_type(8)))  int      v8i;
typedef __attribute__((ext_vector_type(4)))  int      v4i;

// ---------------------------------------------------------------------------
// Small helpers
// ---------------------------------------------------------------------------
__device__ __forceinline__ void topk_insert(float (&tv)[kTopK], int (&ti)[kTopK],
                                            float v, int idx) {
  if (v <= tv[kTopK - 1]) return;
  tv[kTopK - 1] = v;
  ti[kTopK - 1] = idx;
#pragma unroll
  for (int k = kTopK - 1; k > 0; --k) {
    if (tv[k] > tv[k - 1]) {
      float tf = tv[k]; tv[k] = tv[k - 1]; tv[k - 1] = tf;
      int   tq = ti[k]; ti[k] = ti[k - 1]; ti[k - 1] = tq;
    }
  }
}

// Issue one TDM load: 16 x 256 f16 tile, global -> LDS (with 16B/row padding).
// D# layout per CDNA5 ISA ch.8 (group0 128b, group1 256b, groups 2/3 zero=2D).
__device__ __forceinline__ void tdm_load_tile(unsigned long long gaddr,
                                              unsigned int lds_off) {
  v4u g0;
  g0[0] = 1u;                                            // count=1, user mode
  g0[1] = lds_off;                                       // LDS byte address
  g0[2] = (unsigned int)(gaddr & 0xFFFFFFFFull);         // global_addr[31:0]
  g0[3] = (unsigned int)((gaddr >> 32) & 0x01FFFFFFull)  // global_addr[56:32]
          | (2u << 30);                                  // type=2 ("image")
  v8i g1;
  g1[0] = (int)((1u << 16)    // data_size = 2 bytes
              | (1u << 20)    // pad_enable
              | (6u << 22)    // pad_interval: 128 DWORDs (one 512B row)
              | (3u << 25));  // pad_amount: 4 DWORDs (16B)
  g1[1] = (int)(256u << 16);    // tensor_dim0[15:0]=256 (bits 63:48)
  g1[2] = (int)(16384u << 16);  // tensor_dim1[15:0]=16384 (bits 95:80)
  g1[3] = (int)(256u << 16);    // tile_dim0=256 (bits 127:112)
  g1[4] = 16;                   // tile_dim1=16
  g1[5] = 256;                  // tensor_dim0_stride=256 elements
  g1[6] = 0;
  g1[7] = 0;
  v4i gz  = {0, 0, 0, 0};               // groups 2/3 unused (2D tensor)
  v8i gz8 = {0, 0, 0, 0, 0, 0, 0, 0};   // trailing group (6-arg builtin form)
  __builtin_amdgcn_tensor_load_to_lds(g0, g1, gz, gz, gz8, 0);
}

// ---------------------------------------------------------------------------
// Kernel 1: row-wise L2 normalize (f32 -> f16), one wave per 256-elem row.
// ---------------------------------------------------------------------------
__global__ void __launch_bounds__(128)
lmm_normalize_f16(const float* __restrict__ src,
                  _Float16* __restrict__ dst, int nrows) {
  const int row  = blockIdx.x * (blockDim.x >> 5) + (threadIdx.x >> 5);
  const int lane = threadIdx.x & 31;
  if (row >= nrows) return;
  const float* s = src + (size_t)row * kD + lane * 8;
  float v[8];
  float ss = 0.f;
#pragma unroll
  for (int t = 0; t < 8; ++t) { v[t] = s[t]; ss += v[t] * v[t]; }
#pragma unroll
  for (int off = 16; off >= 1; off >>= 1) ss += __shfl_xor(ss, off, 32);
  const float inv = 1.0f / fmaxf(sqrtf(ss), 1e-12f);
  _Float16* d = dst + (size_t)row * kD + lane * 8;
#pragma unroll
  for (int t = 0; t < 8; ++t) d[t] = (_Float16)(v[t] * inv);
}

// ---------------------------------------------------------------------------
// Kernel 2: fused cosine-sim GEMM (f16 WMMA, f32 accum) + per-row top-5.
// Block = 4 waves, each owning a 16-row L tile; one 16-col B tile staged in
// LDS by the Tensor Data Mover is consumed by all 4 waves (4x reuse).
// Wave 0 drives the TDM two tiles ahead (double-buffered LDS stages) and
// gates the block on s_wait_tensorcnt. All 16 ds_load_b128 of a tile are
// issued as one clause into a register array before the 8 back-to-back WMMAs.
// ---------------------------------------------------------------------------
struct TopKState {
  float tv[kTopK];
  int   ti[kTopK];
};

__global__ void __launch_bounds__(128, 1)
lmm_sim_topk(const _Float16* __restrict__ enc_n,
             const _Float16* __restrict__ mem_n,
             float* __restrict__ pvals,
             int*   __restrict__ pidx) {
  __shared__ struct __align__(16) {
    unsigned int stage[2][kStageDW];   // double-buffered B tile (TDM target)
    float simt[kWavesPerBlk][16 * 17]; // per-wave C-tile scratch (padded)
  } sh;

  const int wg    = blockIdx.x;
  const int chunk = wg % kNChunks;
  const int lgrp  = (wg / kNChunks) % (kL / (16 * kWavesPerBlk));
  const int b     = wg / (kNChunks * (kL / (16 * kWavesPerBlk)));
  const int wave  = threadIdx.x >> 5;
  const int lane  = threadIdx.x & 31;
  const int half  = lane >> 4;          // 0 or 1
  const int l15   = lane & 15;
  const int kOff  = half ? 8 : 0;       // per-lane K sub-offset (halves)

  const int lbase = lgrp * (16 * kWavesPerBlk) + wave * 16;
  const int n0    = chunk * kChunk;

  // Preload this wave's A fragments for K=256 (8 steps of K=32): 64 VGPRs.
  const _Float16* Abase = enc_n + (size_t)(b * kL + lbase + l15) * kD;
  v16h areg[8];
#pragma unroll
  for (int ks = 0; ks < 8; ++ks) {
    const _Float16* ap = Abase + ks * 32 + kOff;
    v8h lo = *(const v8h*)ap;
    v8h hi = *(const v8h*)(ap + 16);
#pragma unroll
    for (int t = 0; t < 8; ++t) { areg[ks][t] = lo[t]; areg[ks][t + 8] = hi[t]; }
  }

  TopKState st;
#pragma unroll
  for (int k = 0; k < kTopK; ++k) { st.tv[k] = -3.0e38f; st.ti[k] = 0; }

  const int scanBase = half * 8;        // lanes 0-15: cols 0-7; 16-31: cols 8-15
  const unsigned long long gbase =
      (unsigned long long)(uintptr_t)(mem_n + ((size_t)b * kN + n0) * kD);
  const unsigned int stage_off[2] = {
      (unsigned int)(uintptr_t)(void*)&sh.stage[0][0],
      (unsigned int)(uintptr_t)(void*)&sh.stage[1][0] };

  // Prologue: wave 0 launches TDM for tiles 0 and 1.
  if (wave == 0) {
    tdm_load_tile(gbase, stage_off[0]);
    tdm_load_tile(gbase + 16ull * kD * 2, stage_off[1]);
  }

  for (int t = 0; t < kTiles; ++t) {
    if (wave == 0) {
      if (t + 1 < kTiles) __builtin_amdgcn_s_wait_tensorcnt(1);  // tile t done
      else                __builtin_amdgcn_s_wait_tensorcnt(0);
    }
    __syncthreads();                    // stage[t&1] visible to all waves

    // Pull the whole staged tile into registers (one DS clause)...
    const _Float16* sp =
        (const _Float16*)&sh.stage[t & 1][0] + l15 * (kRowDW * 2) + kOff;
    v16h breg[8];
#pragma unroll
    for (int ks = 0; ks < 8; ++ks) {
      v8h lo = *(const v8h*)(sp + ks * 32);
      v8h hi = *(const v8h*)(sp + ks * 32 + 16);
#pragma unroll
      for (int q = 0; q < 8; ++q) { breg[ks][q] = lo[q]; breg[ks][q + 8] = hi[q]; }
    }
    // ...then 8 back-to-back WMMAs.
    v8f acc = {};
#pragma unroll
    for (int ks = 0; ks < 8; ++ks) {
      acc = __builtin_amdgcn_wmma_f32_16x16x32_f16(
          false, areg[ks], false, breg[ks], (short)0, acc, false, false);
    }

    // Spill C tile to this wave's private scratch; scan; no barrier needed
    // (DS ops are in-order within a wave).
    float* mysim = &sh.simt[wave][0];
    const int mb = half * 8;
    const int nbase = n0 + t * 16;
#pragma unroll
    for (int v = 0; v < 8; ++v) mysim[(mb + v) * 17 + l15] = acc[v];
#pragma unroll
    for (int j = 0; j < 8; ++j) {
      const float val = mysim[l15 * 17 + scanBase + j];
      topk_insert(st.tv, st.ti, val, nbase + scanBase + j);
    }

    __syncthreads();                    // all waves done with stage[t&1]
    if (wave == 0 && t + 2 < kTiles)
      tdm_load_tile(gbase + (unsigned long long)(t + 2) * 16 * kD * 2,
                    stage_off[t & 1]);
  }

  const size_t base =
      ((size_t)(b * kL + lbase + l15) * kNPart + chunk * 2 + half) * kTopK;
#pragma unroll
  for (int k = 0; k < kTopK; ++k) { pvals[base + k] = st.tv[k]; pidx[base + k] = st.ti[k]; }
}

// ---------------------------------------------------------------------------
// Kernel 3: merge 16 partial top-5 lists per row -> true top-5, gather raw
// memory rows, mean, add encoded. One wave per (b,l) row.
// ---------------------------------------------------------------------------
__global__ void __launch_bounds__(32)
lmm_merge_gather(const float* __restrict__ pvals,
                 const int*   __restrict__ pidx,
                 const float* __restrict__ encoded,
                 const float* __restrict__ memory,
                 float* __restrict__ xbuf) {
  const int rowid = blockIdx.x;           // 0 .. B*L-1
  const int b     = rowid / kL;
  const int lane  = threadIdx.x;
  __shared__ int topi[kTopK];

  if (lane == 0) {
    float tv[kTopK];
    int   ti[kTopK];
#pragma unroll
    for (int k = 0; k < kTopK; ++k) { tv[k] = -3.0e38f; ti[k] = 0; }
    const float* pv = pvals + (size_t)rowid * kNPart * kTopK;
    const int*   pi = pidx  + (size_t)rowid * kNPart * kTopK;
    for (int c = 0; c < kNPart * kTopK; ++c) topk_insert(tv, ti, pv[c], pi[c]);
#pragma unroll
    for (int k = 0; k < kTopK; ++k) topi[k] = ti[k];
  }
  __syncthreads();

  const float* mem = memory  + (size_t)b * kN * kD;
  const float* enc = encoded + (size_t)rowid * kD;
  float*       x   = xbuf    + (size_t)rowid * kD;
  const int i0 = topi[0], i1 = topi[1], i2 = topi[2], i3 = topi[3], i4 = topi[4];
#pragma unroll
  for (int t = 0; t < kD / 32; ++t) {
    const int d = lane + t * 32;
    const float s = mem[(size_t)i0 * kD + d] + mem[(size_t)i1 * kD + d] +
                    mem[(size_t)i2 * kD + d] + mem[(size_t)i3 * kD + d] +
                    mem[(size_t)i4 * kD + d];
    x[d] = enc[d] + s * 0.2f;
  }
}

// ---------------------------------------------------------------------------
// Kernel 4: out = x @ W^T + bias, full-precision via V_WMMA_F32_16X16X4_F32.
// One wave per 16x16 output tile; 64 WMMAs over K=256, loads batched 8 ahead.
// ---------------------------------------------------------------------------
__global__ void __launch_bounds__(32, 1)
lmm_linear(const float* __restrict__ xbuf,
           const float* __restrict__ W,
           const float* __restrict__ bias,
           float* __restrict__ out) {
  const int wg   = blockIdx.x;
  const int jt   = wg % (kD / 16);          // output-feature tile
  const int it   = wg / (kD / 16);          // row tile
  const int lane = threadIdx.x;
  const int l15  = lane & 15;
  const int kb   = (lane & 16) ? 2 : 0;

  const float* xr = xbuf + (size_t)(it * 16 + l15) * kD + kb;  // A row m
  const float* wr = W    + (size_t)(jt * 16 + l15) * kD + kb;  // B col n = W row j

  v8f acc = {};
  for (int k0 = 0; k0 < kD; k0 += 32) {
    v2f aa[8], bb[8];
#pragma unroll
    for (int q = 0; q < 8; ++q) {
      aa[q] = *(const v2f*)(xr + k0 + q * 4);
      bb[q] = *(const v2f*)(wr + k0 + q * 4);
    }
#pragma unroll
    for (int q = 0; q < 8; ++q) {
      acc = __builtin_amdgcn_wmma_f32_16x16x4_f32(
          false, aa[q], false, bb[q], (short)0, acc, false, false);
    }
  }

  const int mrow = it * 16 + ((lane & 16) ? 8 : 0);
  const int col  = jt * 16 + l15;
  const float bv = bias[col];
#pragma unroll
  for (int v = 0; v < 8; ++v)
    out[(size_t)(mrow + v) * kD + col] = acc[v] + bv;
}

// ---------------------------------------------------------------------------
// Host launcher. Workspace layout (~41 MB):
//   mem_n f16 [B,N,D] | enc_n f16 [B,L,D] | pvals f32 | pidx i32 | xbuf f32
// ---------------------------------------------------------------------------
extern "C" void kernel_launch(void* const* d_in, const int* in_sizes, int n_in,
                              void* d_out, int out_size, void* d_ws, size_t ws_size,
                              hipStream_t stream) {
  (void)in_sizes; (void)n_in; (void)out_size; (void)ws_size;
  const float* encoded = (const float*)d_in[0];   // [B,L,D]
  const float* memory  = (const float*)d_in[1];   // [B,N,D]
  const float* W       = (const float*)d_in[2];   // [D,D]
  const float* bias    = (const float*)d_in[3];   // [D]
  float* out = (float*)d_out;                     // [B,L,D]

  char* ws = (char*)d_ws;
  _Float16* mem_n = (_Float16*)ws;  ws += (size_t)kB * kN * kD * sizeof(_Float16);
  _Float16* enc_n = (_Float16*)ws;  ws += (size_t)kB * kL * kD * sizeof(_Float16);
  float* pvals    = (float*)ws;     ws += (size_t)kB * kL * kNPart * kTopK * sizeof(float);
  int*   pidx     = (int*)ws;       ws += (size_t)kB * kL * kNPart * kTopK * sizeof(int);
  float* xbuf     = (float*)ws;     // (size_t)kB * kL * kD * sizeof(float)

  // 1) normalize to f16
  lmm_normalize_f16<<<(kB * kN) / 4, 128, 0, stream>>>(memory,  mem_n, kB * kN);
  lmm_normalize_f16<<<(kB * kL) / 4, 128, 0, stream>>>(encoded, enc_n, kB * kL);

  // 2) fused sim GEMM + top-5 (f16 WMMA, TDM-staged B tiles): 512 blocks x 4 waves
  lmm_sim_topk<<<kB * (kL / (16 * kWavesPerBlk)) * kNChunks, 32 * kWavesPerBlk,
                 0, stream>>>(enc_n, mem_n, pvals, pidx);

  // 3) merge partials, gather memory rows, add encoded
  lmm_merge_gather<<<kB * kL, 32, 0, stream>>>(pvals, pidx, encoded, memory, xbuf);

  // 4) final linear in f32 WMMA: 4096 tiles
  lmm_linear<<<(kB * kL / 16) * (kD / 16), 32, 0, stream>>>(xbuf, W, bias, out);
}
